// MambaBlock_23450521436677
// MI455X (gfx1250) — matrix operand order repaired and checked by reading, not compiled, not run on hardware
//
#include <hip/hip_runtime.h>
#include <hip/hip_bf16.h>
#include <math.h>

#define DIM      768
#define D_STATE  16
#define D_CONV   4
#define DT_RANK  32
#define D_INNER  1536   /* EXPAND * DIM */
#define BATCH    2
#define SEQ      1024
#define MROWS    (BATCH * SEQ)   /* 2048 */

#define BM 128
#define BN 64
#define BK 32

typedef __attribute__((ext_vector_type(16))) __bf16 v16bf;
typedef __attribute__((ext_vector_type(8)))  __bf16 v8bf;
typedef __attribute__((ext_vector_type(8)))  float  v8f;
typedef __attribute__((ext_vector_type(4)))  int    v4i;

// CDNA5 async global->LDS path (ASYNCcnt), with sync fallback for safety.
// Probe-learned signature: (global int4* src, lds int4* dst, imm offset, imm cpol)
#if defined(__AMDGCN__) && __has_builtin(__builtin_amdgcn_global_load_async_to_lds_b128)
  #define USE_ASYNC_LDS 1
  typedef __attribute__((address_space(1))) v4i as1_v4i;
  typedef __attribute__((address_space(3))) v4i as3_v4i;
  #define ASYNC_WAIT_3() asm volatile("s_wait_asynccnt 0x3" ::: "memory")
  #define ASYNC_WAIT_0() asm volatile("s_wait_asynccnt 0x0" ::: "memory")
#else
  #define USE_ASYNC_LDS 0
  #define ASYNC_WAIT_3()
  #define ASYNC_WAIT_0()
#endif

static __device__ __forceinline__ __bf16 f2bf(float f) {
    unsigned u = __builtin_bit_cast(unsigned, f);
    unsigned r = u + 0x7FFFu + ((u >> 16) & 1u);   // round-to-nearest-even
    unsigned short h = (unsigned short)(r >> 16);
    return __builtin_bit_cast(__bf16, h);
}

static __device__ __forceinline__ float siluf(float v) {
    return v / (1.0f + __expf(-v));
}

// ---------------------------------------------------------------------------
// Elementwise f32 -> bf16 cast
// ---------------------------------------------------------------------------
__global__ void cast_f32_bf16_kernel(const float* __restrict__ in,
                                     __bf16* __restrict__ out, int n) {
    int i = blockIdx.x * blockDim.x + threadIdx.x;
    if (i < n) out[i] = f2bf(in[i]);
}

// dt_lr slice (first 32 of 64 cols of x_dbl) -> bf16, row stride 64 -> 32
__global__ void dtlr_cast_kernel(const float* __restrict__ x_dbl,
                                 __bf16* __restrict__ out) {
    int i = blockIdx.x * blockDim.x + threadIdx.x;
    if (i < MROWS * DT_RANK) {
        int m = i >> 5, r = i & 31;
        out[i] = f2bf(x_dbl[m * (DT_RANK + 2 * D_STATE) + r]);
    }
}

// ---------------------------------------------------------------------------
// Tiled bf16 WMMA GEMM:  C[M,N] = A[M,K] * Bw[N,K]^T  (+ optional residual)
//
// Block tile 128(M) x 64(N), K-step 32, double-buffered LDS filled with
// GLOBAL_LOAD_ASYNC_TO_LDS_B128 (3 x 16B per thread per slab, ASYNCcnt-
// pipelined). 8 waves: wave w -> (wm = w>>1 in 0..3, wn = w&1); each wave owns
// a 32x32 output tile = 4 accumulators -> 4 v_wmma_f32_16x16x32_bf16 per slab.
// LDS layout makes per-lane fragment loads contiguous 16B chunks matching the
// ISA bf16 A(16x32) / B(32x16) VGPR striping.
// ---------------------------------------------------------------------------
__global__ __launch_bounds__(256)
void gemm_bf16_wmma_kernel(const __bf16* __restrict__ A,
                           const __bf16* __restrict__ Bw,
                           float* __restrict__ Cout,
                           const float* __restrict__ residual,
                           int M, int N, int K) {
    __shared__ __bf16 sA[2][BM * BK];   // [m][k] row-major, stride 32
    __shared__ __bf16 sB[2][BN * BK];   // [n][k] row-major, stride 32

    const int tid  = threadIdx.x;
    const int lane = tid & 31;
    const int wave = tid >> 5;
    const int wm   = wave >> 1;          // 0..3 -> 32-row slab
    const int wn   = wave & 1;           // 0..1 -> 32-col slab
    const int m0   = blockIdx.y * BM;
    const int n0   = blockIdx.x * BN;

    const int half = lane >> 4;          // 0 = lanes 0-15, 1 = lanes 16-31
    const int lrow = lane & 15;

    const int f0 = tid * 8;              // flat bf16 offset for staging (16B/thread)
    const int srow = f0 >> 5;            // 0..63
    const int scol = f0 & 31;            // 0,8,16,24

    v8f acc00 = {}, acc01 = {}, acc10 = {}, acc11 = {};

    auto stage = [&](int k0, int buf) {
#if USE_ASYNC_LDS
        __builtin_amdgcn_global_load_async_to_lds_b128(
            (as1_v4i*)&A[(size_t)(m0 + srow) * K + k0 + scol],
            (as3_v4i*)&sA[buf][f0], 0, 0);
        __builtin_amdgcn_global_load_async_to_lds_b128(
            (as1_v4i*)&A[(size_t)(m0 + 64 + srow) * K + k0 + scol],
            (as3_v4i*)&sA[buf][2048 + f0], 0, 0);
        __builtin_amdgcn_global_load_async_to_lds_b128(
            (as1_v4i*)&Bw[(size_t)(n0 + srow) * K + k0 + scol],
            (as3_v4i*)&sB[buf][f0], 0, 0);
#else
        *reinterpret_cast<v8bf*>(&sA[buf][f0]) =
            *reinterpret_cast<const v8bf*>(&A[(size_t)(m0 + srow) * K + k0 + scol]);
        *reinterpret_cast<v8bf*>(&sA[buf][2048 + f0]) =
            *reinterpret_cast<const v8bf*>(&A[(size_t)(m0 + 64 + srow) * K + k0 + scol]);
        *reinterpret_cast<v8bf*>(&sB[buf][f0]) =
            *reinterpret_cast<const v8bf*>(&Bw[(size_t)(n0 + srow) * K + k0 + scol]);
#endif
    };

    // A fragment: lanes 0-15 K=[kb..kb+7, kb+16..kb+23] with kb=half*8
    auto fragA = [&](const __bf16* base) {
        v8bf lo = *reinterpret_cast<const v8bf*>(base);
        v8bf hi = *reinterpret_cast<const v8bf*>(base + 16);
        v16bf r;
        #pragma unroll
        for (int i = 0; i < 8; ++i) { r[i] = lo[i]; r[i + 8] = hi[i]; }
        return r;
    };
    // B fragment: same K order as A: K=[kb..kb+7, kb+16..kb+23] with kb=half*8
    auto fragB = [&](const __bf16* base) {
        v8bf lo = *reinterpret_cast<const v8bf*>(base);
        v8bf hi = *reinterpret_cast<const v8bf*>(base + 16);
        v16bf r;
        #pragma unroll
        for (int i = 0; i < 8; ++i) { r[i] = lo[i]; r[i + 8] = hi[i]; }
        return r;
    };

    stage(0, 0);
    int buf = 0;
    for (int k0 = 0; k0 < K; k0 += BK, buf ^= 1) {
        if (k0 + BK < K) {
            stage(k0 + BK, buf ^ 1);   // prefetch next slab into other buffer
            ASYNC_WAIT_3();            // wait for current slab only
        } else {
            ASYNC_WAIT_0();
        }
        __syncthreads();

        v16bf a0 = fragA(&sA[buf][(wm * 32 + lrow) * 32 + half * 8]);
        v16bf a1 = fragA(&sA[buf][(wm * 32 + 16 + lrow) * 32 + half * 8]);
        v16bf b0 = fragB(&sB[buf][(wn * 32 + lrow) * 32 + half * 8]);
        v16bf b1 = fragB(&sB[buf][(wn * 32 + 16 + lrow) * 32 + half * 8]);

        acc00 = __builtin_amdgcn_wmma_f32_16x16x32_bf16(false, a0, false, b0, (short)0, acc00, false, false);
        acc01 = __builtin_amdgcn_wmma_f32_16x16x32_bf16(false, a0, false, b1, (short)0, acc01, false, false);
        acc10 = __builtin_amdgcn_wmma_f32_16x16x32_bf16(false, a1, false, b0, (short)0, acc10, false, false);
        acc11 = __builtin_amdgcn_wmma_f32_16x16x32_bf16(false, a1, false, b1, (short)0, acc11, false, false);

        __syncthreads();
    }

    // C/D layout: VGPR i -> M = i + 8*half, N = lrow (within each 16x16 tile)
    const v8f* accs[2][2] = {{&acc00, &acc01}, {&acc10, &acc11}};
    #pragma unroll
    for (int i = 0; i < 2; ++i) {
        #pragma unroll
        for (int j = 0; j < 2; ++j) {
            const int nn = n0 + wn * 32 + j * 16 + lrow;
            #pragma unroll
            for (int e = 0; e < 8; ++e) {
                const int mm = m0 + wm * 32 + i * 16 + e + half * 8;
                const size_t off = (size_t)mm * N + nn;
                float v = (*accs[i][j])[e];
                if (residual) v += residual[off];
                Cout[off] = v;
            }
        }
    }
}

// ---------------------------------------------------------------------------
// Causal depthwise conv (width 4) + bias + SiLU over xc = xz[:, :D_INNER].
// ---------------------------------------------------------------------------
__global__ void conv_silu_kernel(const float* __restrict__ xz,
                                 const float* __restrict__ W_conv,  // [D_INNER][4]
                                 const float* __restrict__ b_conv,  // [D_INNER]
                                 float* __restrict__ xc_f32,
                                 __bf16* __restrict__ xc_bf) {
    int idx = blockIdx.x * blockDim.x + threadIdx.x;
    if (idx >= MROWS * D_INNER) return;
    const int d = idx % D_INNER;
    const int l = (idx / D_INNER) % SEQ;
    const int b = idx / (D_INNER * SEQ);

    float acc = b_conv[d];
    #pragma unroll
    for (int j = 0; j < D_CONV; ++j) {
        const int li = l - (D_CONV - 1) + j;
        if (li >= 0)
            acc += xz[((size_t)(b * SEQ + li)) * (2 * D_INNER) + d] * W_conv[d * D_CONV + j];
    }
    const float s = siluf(acc);
    const size_t off = (size_t)(b * SEQ + l) * D_INNER + d;
    xc_f32[off] = s;
    xc_bf[off]  = f2bf(s);
}

// ---------------------------------------------------------------------------
// Selective scan: one thread per (batch, channel); 16 states in registers.
// B/C vectors (shared by all channels of a (b,l)) staged through LDS.
// ---------------------------------------------------------------------------
__global__ void mamba_scan_kernel(const float* __restrict__ xz,     // [M][3072]
                                  const float* __restrict__ xc,     // [M][1536]
                                  const float* __restrict__ x_dbl,  // [M][64]
                                  const float* __restrict__ dt_pre, // [M][1536]
                                  const float* __restrict__ b_dt,   // [1536]
                                  const float* __restrict__ A_log,  // [1536][16]
                                  const float* __restrict__ Dp,     // [1536]
                                  __bf16* __restrict__ y_bf) {      // [M][1536]
    __shared__ float sB[D_STATE];
    __shared__ float sC[D_STATE];

    const int gid = blockIdx.x * blockDim.x + threadIdx.x;
    const int d = gid % D_INNER;
    const int b = gid / D_INNER;      // 6 blocks per batch -> whole block shares b

    float Ac[D_STATE], h[D_STATE];
    #pragma unroll
    for (int s = 0; s < D_STATE; ++s) {
        Ac[s] = -__expf(A_log[d * D_STATE + s]);
        h[s] = 0.0f;
    }
    const float bdt2 = 2.0f * b_dt[d];
    const float Dd = Dp[d];

    for (int l = 0; l < SEQ; ++l) {
        const int row = b * SEQ + l;
        if (threadIdx.x < D_STATE)
            sB[threadIdx.x] = x_dbl[row * 64 + DT_RANK + threadIdx.x];
        else if (threadIdx.x < 2 * D_STATE)
            sC[threadIdx.x - D_STATE] = x_dbl[row * 64 + DT_RANK + D_STATE + (threadIdx.x - D_STATE)];
        __syncthreads();

        const float dtp = dt_pre[(size_t)row * D_INNER + d] + bdt2;
        const float dt  = (dtp > 20.0f) ? dtp : log1pf(__expf(dtp));
        const float u   = xc[(size_t)row * D_INNER + d];
        const float z   = xz[(size_t)row * (2 * D_INNER) + D_INNER + d];

        float y = 0.0f;
        const float dtu = dt * u;
        #pragma unroll
        for (int s = 0; s < D_STATE; ++s) {
            h[s] = __expf(dt * Ac[s]) * h[s] + dtu * sB[s];
            y += h[s] * sC[s];
        }
        y = (y + u * Dd) * siluf(z);
        y_bf[(size_t)row * D_INNER + d] = f2bf(y);
        __syncthreads();
    }
}

// ---------------------------------------------------------------------------
// Host-side launch
// ---------------------------------------------------------------------------
extern "C" void kernel_launch(void* const* d_in, const int* in_sizes, int n_in,
                              void* d_out, int out_size, void* d_ws, size_t ws_size,
                              hipStream_t stream) {
    (void)in_sizes; (void)n_in; (void)out_size; (void)ws_size;
    const float* x      = (const float*)d_in[0];
    const float* W_in   = (const float*)d_in[1];
    const float* W_conv = (const float*)d_in[2];
    const float* b_conv = (const float*)d_in[3];
    const float* W_x    = (const float*)d_in[4];
    const float* W_dt   = (const float*)d_in[5];
    const float* b_dt   = (const float*)d_in[6];
    const float* A_log  = (const float*)d_in[7];
    const float* Dp     = (const float*)d_in[8];
    const float* W_out  = (const float*)d_in[9];
    float* out = (float*)d_out;

    char* ws = (char*)d_ws;
    size_t off = 0;
    auto alloc = [&](size_t bytes) -> void* {
        void* p = ws + off;
        off = (off + bytes + 255) & ~(size_t)255;
        return p;
    };

    __bf16* x_bf     = (__bf16*)alloc((size_t)MROWS * DIM * 2);
    __bf16* W_in_bf  = (__bf16*)alloc((size_t)2 * D_INNER * DIM * 2);
    __bf16* W_x_bf   = (__bf16*)alloc((size_t)(DT_RANK + 2 * D_STATE) * D_INNER * 2);
    __bf16* W_dt_bf  = (__bf16*)alloc((size_t)D_INNER * DT_RANK * 2);
    __bf16* W_out_bf = (__bf16*)alloc((size_t)DIM * D_INNER * 2);
    float*  xz       = (float*) alloc((size_t)MROWS * 2 * D_INNER * 4);
    float*  xc_f32   = (float*) alloc((size_t)MROWS * D_INNER * 4);
    __bf16* xc_bf    = (__bf16*)alloc((size_t)MROWS * D_INNER * 2);
    float*  x_dbl    = (float*) alloc((size_t)MROWS * (DT_RANK + 2 * D_STATE) * 4);
    __bf16* dtlr_bf  = (__bf16*)alloc((size_t)MROWS * DT_RANK * 2);
    float*  dt_pre   = (float*) alloc((size_t)MROWS * D_INNER * 4);
    __bf16* y_bf     = (__bf16*)alloc((size_t)MROWS * D_INNER * 2);

    const int T = 256;
    auto cdiv = [](int a, int b) { return (a + b - 1) / b; };

    // 1) casts to bf16
    cast_f32_bf16_kernel<<<cdiv(MROWS * DIM, T), T, 0, stream>>>(x, x_bf, MROWS * DIM);
    cast_f32_bf16_kernel<<<cdiv(2 * D_INNER * DIM, T), T, 0, stream>>>(W_in, W_in_bf, 2 * D_INNER * DIM);
    cast_f32_bf16_kernel<<<cdiv((DT_RANK + 2 * D_STATE) * D_INNER, T), T, 0, stream>>>(W_x, W_x_bf, (DT_RANK + 2 * D_STATE) * D_INNER);
    cast_f32_bf16_kernel<<<cdiv(D_INNER * DT_RANK, T), T, 0, stream>>>(W_dt, W_dt_bf, D_INNER * DT_RANK);
    cast_f32_bf16_kernel<<<cdiv(DIM * D_INNER, T), T, 0, stream>>>(W_out, W_out_bf, DIM * D_INNER);

    // 2) xz = x @ W_in^T   (M=2048, N=3072, K=768)
    gemm_bf16_wmma_kernel<<<dim3((2 * D_INNER) / BN, MROWS / BM), T, 0, stream>>>(
        x_bf, W_in_bf, xz, nullptr, MROWS, 2 * D_INNER, DIM);

    // 3) depthwise causal conv + bias + SiLU
    conv_silu_kernel<<<cdiv(MROWS * D_INNER, T), T, 0, stream>>>(
        xz, W_conv, b_conv, xc_f32, xc_bf);

    // 4) x_dbl = xc @ W_x^T   (M=2048, N=64, K=1536)
    gemm_bf16_wmma_kernel<<<dim3((DT_RANK + 2 * D_STATE) / BN, MROWS / BM), T, 0, stream>>>(
        xc_bf, W_x_bf, x_dbl, nullptr, MROWS, DT_RANK + 2 * D_STATE, D_INNER);

    // 5) dt_pre = dt_lr @ W_dt^T   (M=2048, N=1536, K=32)
    dtlr_cast_kernel<<<cdiv(MROWS * DT_RANK, T), T, 0, stream>>>(x_dbl, dtlr_bf);
    gemm_bf16_wmma_kernel<<<dim3(D_INNER / BN, MROWS / BM), T, 0, stream>>>(
        dtlr_bf, W_dt_bf, dt_pre, nullptr, MROWS, D_INNER, DT_RANK);

    // 6) selective scan + gating -> y (bf16)
    mamba_scan_kernel<<<(BATCH * D_INNER) / T, T, 0, stream>>>(
        xz, xc_f32, x_dbl, dt_pre, b_dt, A_log, Dp, y_bf);

    // 7) out = y @ W_out^T + x   (M=2048, N=768, K=1536), residual fused
    gemm_bf16_wmma_kernel<<<dim3(DIM / BN, MROWS / BM), T, 0, stream>>>(
        y_bf, W_out_bf, out, x, MROWS, DIM, D_INNER);
}
